// KGCompletionGNN_32976758898977
// MI455X (gfx1250) — matrix operand order repaired and checked by reading, not compiled, not run on hardware
//
#include <hip/hip_runtime.h>

#define NN 50000
#define MM 200000
#define DD 256
#define TE 32            // edges per workgroup tile
#define NKT 24           // K tiles of 32 (K = 768)
#define TILE_ELEMS (2*8*32*16)   // [rowtile][kt8][lane][16] = 8192 bf16 per segment
#define PWSZ (16L*24*32*16)      // packed weight elems per matrix (196608)
#define SLOPE 0.01f
#define LNEPS 1e-5f

typedef __attribute__((ext_vector_type(16))) __bf16 bf16x16;
typedef __attribute__((ext_vector_type(8)))  float  f32x8;

__device__ __forceinline__ f32x8 wmma_bf16(bf16x16 a, bf16x16 b, f32x8 c) {
  return __builtin_amdgcn_wmma_f32_16x16x32_bf16(false, a, false, b, (short)0, c,
                                                 false, false);
}
__device__ __forceinline__ float leaky(float x) { return x >= 0.f ? x : SLOPE * x; }

// ---------------------------------------------------------------------------
// Pack weights into WMMA B-fragment order (bf16).
// Fragment element j of lane L (n = ct*16 + (L&15), khalf = L>>4) is
// W'[kt*32 + khalf*8 + (j&7) + (j>>3)*16][n].  mode 0: message combine (src 1024x256):
//   k<256 -> W[k]+W[k+512]; k<512 -> W[k]+W[k+256]; else W[k+256]
// mode 1: identity (src 768x256).
__global__ __launch_bounds__(256)
void pack_kernel(const float* __restrict__ W, __bf16* __restrict__ out, int mode) {
  long id = (long)blockIdx.x * 256 + threadIdx.x;
  if (id >= PWSZ) return;
  int j    = (int)(id & 15);
  long r   = id >> 4;
  int lane = (int)(r & 31);  r >>= 5;
  int kt   = (int)(r % 24);
  int ct   = (int)(r / 24);
  int k = kt * 32 + (lane >> 4) * 8 + (j & 7) + (j >> 3) * 16;
  int n = ct * 16 + (lane & 15);
  float v;
  if (mode == 0) {
    if      (k < 256) v = W[(long)k*DD + n] + W[(long)(k+512)*DD + n];
    else if (k < 512) v = W[(long)k*DD + n] + W[(long)(k+256)*DD + n];
    else              v = W[(long)(k+256)*DD + n];
  } else {
    v = W[(long)k*DD + n];
  }
  out[id] = (__bf16)v;
}

// ---------------------------------------------------------------------------
__global__ __launch_bounds__(256)
void count_kernel(const long long* __restrict__ ht, float* __restrict__ cnt) {
  int e = blockIdx.x * 256 + threadIdx.x;
  if (e < MM) {
    atomicAdd(&cnt[(int)ht[2*e + 0]], 1.0f);
    atomicAdd(&cnt[(int)ht[2*e + 1]], 1.0f);
  }
}

// ---------------------------------------------------------------------------
// Fused message GEMM + scatter-add.  blockIdx.y = direction.
// A (32 edges x 768) = [Hgather | E | Hgather*E], W' packed 768x256.
__global__ __launch_bounds__(256)
void msg_kernel(const float* __restrict__ H, const float* __restrict__ E,
                const long long* __restrict__ ht,
                const __bf16* __restrict__ pWf, const float* __restrict__ bf,
                const __bf16* __restrict__ pWb, const float* __restrict__ bb,
                float* __restrict__ sums)
{
  const int dir = blockIdx.y;
  const long e0 = (long)blockIdx.x * TE;
  const __bf16* __restrict__ pW   = dir ? pWb : pWf;
  const float*  __restrict__ bias = dir ? bb  : bf;

  __shared__ __bf16 tiles[3][TILE_ELEMS];
  __shared__ int idxG[TE], idxS[TE];

  const int t = threadIdx.x;
  if (t < TE) {
    long long a = ht[(e0 + t) * 2 + 0];
    long long c = ht[(e0 + t) * 2 + 1];
    idxG[t] = dir ? (int)c : (int)a;
    idxS[t] = dir ? (int)a : (int)c;
  }
  __syncthreads();

  // Loader: this thread owns column t of the tile; store in A-fragment order.
  {
    const int kt      = t >> 5;
    const int w       = t & 31;
    const int khalf   = (w >> 3) & 1;
    const int fragpos = (w & 7) + ((w >= 16) ? 8 : 0);
    const int lane0   = khalf * 16;
#pragma unroll 4
    for (int r = 0; r < TE; ++r) {
      float h = H[(long)idxG[r] * DD + t];
      float e = E[(e0 + r) * DD + t];
      int off = (((r >> 4) * 8 + kt) * 32 + (lane0 + (r & 15))) * 16 + fragpos;
      tiles[0][off] = (__bf16)h;
      tiles[1][off] = (__bf16)e;
      tiles[2][off] = (__bf16)(h * e);
    }
  }
  __syncthreads();

  const int wv = t >> 5, lane = t & 31;
  f32x8 acc[2][2] = {};
  const long ct0 = 2 * wv, ct1 = 2 * wv + 1;
  for (int kt = 0; kt < NKT; ++kt) {
    const int seg = kt >> 3, kt8 = kt & 7;
    bf16x16 b0 = *(const bf16x16*)(pW + ((ct0 * NKT + kt) * 32 + lane) * 16);
    bf16x16 b1 = *(const bf16x16*)(pW + ((ct1 * NKT + kt) * 32 + lane) * 16);
    bf16x16 a0 = *(const bf16x16*)(&tiles[seg][((0 * 8 + kt8) * 32 + lane) * 16]);
    bf16x16 a1 = *(const bf16x16*)(&tiles[seg][((1 * 8 + kt8) * 32 + lane) * 16]);
    acc[0][0] = wmma_bf16(a0, b0, acc[0][0]);
    acc[0][1] = wmma_bf16(a0, b1, acc[0][1]);
    acc[1][0] = wmma_bf16(a1, b0, acc[1][0]);
    acc[1][1] = wmma_bf16(a1, b1, acc[1][1]);
  }

  const int khalfC = lane >> 4, nloc = lane & 15;
#pragma unroll
  for (int c = 0; c < 2; ++c) {
    const int n = (2 * wv + c) * 16 + nloc;
    const float bv = bias[n];
#pragma unroll
    for (int rt = 0; rt < 2; ++rt) {
#pragma unroll
      for (int v = 0; v < 8; ++v) {
        const int m = rt * 16 + v + 8 * khalfC;
        atomicAdd(&sums[(long)idxS[m] * DD + n], acc[rt][c][v] + bv);
      }
    }
  }
}

// ---------------------------------------------------------------------------
// Node update: agg = sums / max(cnt,1); H' = layernorm(leaky(agg) + H).
__global__ __launch_bounds__(256)
void node_kernel(const float* __restrict__ sums, const float* __restrict__ cnt,
                 const float* __restrict__ Hin, const float* __restrict__ g,
                 const float* __restrict__ bln, float* __restrict__ Hout)
{
  const int wv = threadIdx.x >> 5, lane = threadIdx.x & 31;
  const long row = (long)blockIdx.x * 8 + wv;
  if (row >= NN) return;
  const float inv = 1.0f / fmaxf(cnt[row], 1.0f);
  float x[8];
  float mu = 0.f;
#pragma unroll
  for (int j = 0; j < 8; ++j) {
    const int col = lane + j * 32;
    float a = sums[row * DD + col] * inv;
    x[j] = leaky(a) + Hin[row * DD + col];
    mu += x[j];
  }
  for (int o = 16; o > 0; o >>= 1) mu += __shfl_xor(mu, o, 32);
  mu *= (1.0f / 256.0f);
  float var = 0.f;
#pragma unroll
  for (int j = 0; j < 8; ++j) { float d = x[j] - mu; var += d * d; }
  for (int o = 16; o > 0; o >>= 1) var += __shfl_xor(var, o, 32);
  var *= (1.0f / 256.0f);
  const float rs = rsqrtf(var + LNEPS);
#pragma unroll
  for (int j = 0; j < 8; ++j) {
    const int col = lane + j * 32;
    Hout[row * DD + col] = (x[j] - mu) * rs * g[col] + bln[col];
  }
}

// ---------------------------------------------------------------------------
// Edge update: E' = layernorm(leaky([Hh|E|Ht]@W + b) + E).  In-place safe.
__global__ __launch_bounds__(256)
void edge_kernel(const float* __restrict__ H, const float* __restrict__ Ein,
                 const long long* __restrict__ ht,
                 const __bf16* __restrict__ pW, const float* __restrict__ bias,
                 const float* __restrict__ g, const float* __restrict__ bln,
                 float* __restrict__ Eout)
{
  const long e0 = (long)blockIdx.x * TE;
  __shared__ alignas(32) char shraw[3 * TILE_ELEMS * 2];   // 49152 bytes
  __bf16 (*tiles)[TILE_ELEMS] = reinterpret_cast<__bf16(*)[TILE_ELEMS]>(shraw);
  float (*stage)[264]         = reinterpret_cast<float(*)[264]>(shraw);
  __shared__ int idxH[TE], idxT[TE];

  const int t = threadIdx.x;
  if (t < TE) {
    idxH[t] = (int)ht[(e0 + t) * 2 + 0];
    idxT[t] = (int)ht[(e0 + t) * 2 + 1];
  }
  __syncthreads();
  {
    const int kt      = t >> 5;
    const int w       = t & 31;
    const int khalf   = (w >> 3) & 1;
    const int fragpos = (w & 7) + ((w >= 16) ? 8 : 0);
    const int lane0   = khalf * 16;
#pragma unroll 4
    for (int r = 0; r < TE; ++r) {
      float hh  = H[(long)idxH[r] * DD + t];
      float ee  = Ein[(e0 + r) * DD + t];
      float htl = H[(long)idxT[r] * DD + t];
      int off = (((r >> 4) * 8 + kt) * 32 + (lane0 + (r & 15))) * 16 + fragpos;
      tiles[0][off] = (__bf16)hh;
      tiles[1][off] = (__bf16)ee;
      tiles[2][off] = (__bf16)htl;
    }
  }
  __syncthreads();

  const int wv = t >> 5, lane = t & 31;
  f32x8 acc[2][2] = {};
  const long ct0 = 2 * wv, ct1 = 2 * wv + 1;
  for (int kt = 0; kt < NKT; ++kt) {
    const int seg = kt >> 3, kt8 = kt & 7;
    bf16x16 b0 = *(const bf16x16*)(pW + ((ct0 * NKT + kt) * 32 + lane) * 16);
    bf16x16 b1 = *(const bf16x16*)(pW + ((ct1 * NKT + kt) * 32 + lane) * 16);
    bf16x16 a0 = *(const bf16x16*)(&tiles[seg][((0 * 8 + kt8) * 32 + lane) * 16]);
    bf16x16 a1 = *(const bf16x16*)(&tiles[seg][((1 * 8 + kt8) * 32 + lane) * 16]);
    acc[0][0] = wmma_bf16(a0, b0, acc[0][0]);
    acc[0][1] = wmma_bf16(a0, b1, acc[0][1]);
    acc[1][0] = wmma_bf16(a1, b0, acc[1][0]);
    acc[1][1] = wmma_bf16(a1, b1, acc[1][1]);
  }
  __syncthreads();                     // tiles no longer needed; alias as stage

  const int khalfC = lane >> 4, nloc = lane & 15;
#pragma unroll
  for (int c = 0; c < 2; ++c) {
    const int n = (2 * wv + c) * 16 + nloc;
    const float bv = bias[n];
#pragma unroll
    for (int rt = 0; rt < 2; ++rt) {
#pragma unroll
      for (int v = 0; v < 8; ++v) {
        const int m = rt * 16 + v + 8 * khalfC;
        stage[m][n] = leaky(acc[rt][c][v] + bv);
      }
    }
  }
  __syncthreads();

  // Row-wise residual + layernorm; wave wv owns rows 4wv..4wv+3.
  for (int rr = 0; rr < 4; ++rr) {
    const int row = wv * 4 + rr;
    float x[8];
    float mu = 0.f;
#pragma unroll
    for (int j = 0; j < 8; ++j) {
      const int col = lane * 8 + j;
      x[j] = stage[row][col] + Ein[(e0 + row) * DD + col];
      mu += x[j];
    }
    for (int o = 16; o > 0; o >>= 1) mu += __shfl_xor(mu, o, 32);
    mu *= (1.0f / 256.0f);
    float var = 0.f;
#pragma unroll
    for (int j = 0; j < 8; ++j) { float d = x[j] - mu; var += d * d; }
    for (int o = 16; o > 0; o >>= 1) var += __shfl_xor(var, o, 32);
    var *= (1.0f / 256.0f);
    const float rs = rsqrtf(var + LNEPS);
#pragma unroll
    for (int j = 0; j < 8; ++j) {
      const int col = lane * 8 + j;
      Eout[(e0 + row) * DD + col] = (x[j] - mu) * rs * g[col] + bln[col];
    }
  }
}

// ---------------------------------------------------------------------------
extern "C" void kernel_launch(void* const* d_in, const int* in_sizes, int n_in,
                              void* d_out, int out_size, void* d_ws, size_t ws_size,
                              hipStream_t stream)
{
  const float*     H0 = (const float*)d_in[0];
  const float*     E0 = (const float*)d_in[1];
  const long long* ht = (const long long*)d_in[2];

  float* out  = (float*)d_out;
  float* Hbuf = out;                       // rows [0, N)
  float* Ebuf = out + (long)NN * DD;       // rows [N, N+M)

  char*  ws   = (char*)d_ws;
  float* sums = (float*)ws;                          // N*256 f32
  float* cnt  = sums + (long)NN * DD;                // N f32
  long   pwoff = ((((long)NN * DD + NN) * 4) + 255) & ~255L;
  __bf16* pw   = (__bf16*)(ws + pwoff);              // 6 * PWSZ bf16
  __bf16* pwf0 = pw;
  __bf16* pwb0 = pw + 1 * PWSZ;
  __bf16* pew0 = pw + 2 * PWSZ;
  __bf16* pwf1 = pw + 3 * PWSZ;
  __bf16* pwb1 = pw + 4 * PWSZ;
  __bf16* pew1 = pw + 5 * PWSZ;

  hipMemsetAsync(sums, 0, ((long)NN * DD + NN) * sizeof(float), stream);

  pack_kernel<<<768, 256, 0, stream>>>((const float*)d_in[3],  pwf0, 0);
  pack_kernel<<<768, 256, 0, stream>>>((const float*)d_in[5],  pwb0, 0);
  pack_kernel<<<768, 256, 0, stream>>>((const float*)d_in[9],  pew0, 1);
  pack_kernel<<<768, 256, 0, stream>>>((const float*)d_in[13], pwf1, 0);
  pack_kernel<<<768, 256, 0, stream>>>((const float*)d_in[15], pwb1, 0);
  pack_kernel<<<768, 256, 0, stream>>>((const float*)d_in[19], pew1, 1);
  count_kernel<<<(MM + 255) / 256, 256, 0, stream>>>(ht, cnt);

  // ---- layer 0 ----
  msg_kernel<<<dim3(MM / TE, 2), 256, 0, stream>>>(
      H0, E0, ht, pwf0, (const float*)d_in[4], pwb0, (const float*)d_in[6], sums);
  node_kernel<<<NN / 8, 256, 0, stream>>>(
      sums, cnt, H0, (const float*)d_in[7], (const float*)d_in[8], Hbuf);
  edge_kernel<<<MM / TE, 256, 0, stream>>>(
      Hbuf, E0, ht, pew0, (const float*)d_in[10],
      (const float*)d_in[11], (const float*)d_in[12], Ebuf);

  // ---- layer 1 ----
  hipMemsetAsync(sums, 0, (long)NN * DD * sizeof(float), stream);
  msg_kernel<<<dim3(MM / TE, 2), 256, 0, stream>>>(
      Hbuf, Ebuf, ht, pwf1, (const float*)d_in[14], pwb1, (const float*)d_in[16], sums);
  node_kernel<<<NN / 8, 256, 0, stream>>>(
      sums, cnt, Hbuf, (const float*)d_in[17], (const float*)d_in[18], Hbuf);
  edge_kernel<<<MM / TE, 256, 0, stream>>>(
      Hbuf, Ebuf, ht, pew1, (const float*)d_in[20],
      (const float*)d_in[21], (const float*)d_in[22], Ebuf);
}
